// CTMHead_19799799235213
// MI455X (gfx1250) — compile-verified, matrix operand order checked by reading, not run on hardware
//
#include <hip/hip_runtime.h>

// ---------------------------------------------------------------------------
// CTM head for MI455X (gfx1250, wave32, WMMA).
// Simplifications proven from the reference:
//  * kv length is 1 -> softmax == 1 -> attn is constant across iterations and
//    independent of q (q_w/wq/wk and the whole action-sync chain are dead).
//  * only certs[-1] is returned -> sync_o needed only at the end (== sync_final)
//  * ba/bb are token-independent scalar recurrences -> closed form at the end.
// GEMM strategy: weights pre-transposed+converted to bf16 [N,K] once per call
// (head_w2 bf16 = 77MB -> L2-resident across the 8 M-blocks), activations
// converted f32->bf16 while staging to LDS.  128x128 block tile, 8 wave32
// waves, each wave 32x64 = 8 v_wmma_f32_16x16x32_bf16 per K-step.  B tiles go
// global->LDS via the gfx1250 async-to-LDS path (ASYNCcnt) when available.
// ---------------------------------------------------------------------------

typedef __attribute__((ext_vector_type(16))) __bf16 v16bf;
typedef __attribute__((ext_vector_type(8)))  float  v8f;
typedef __attribute__((__vector_size__(4 * sizeof(int)))) int v4i;

#define TILE_M 128
#define TILE_N 128
#define TILE_K 32
#define LDS_LD (TILE_K + 8)   // even -> bf16 pairs stay dword aligned; 80B rows

#if __has_builtin(__builtin_amdgcn_global_load_async_to_lds_b128) && \
    __has_builtin(__builtin_amdgcn_s_wait_asynccnt)
#define USE_ASYNC_LDS 1
typedef __attribute__((address_space(1))) v4i* gv4i_ptr;   // global
typedef __attribute__((address_space(3))) v4i* lv4i_ptr;   // LDS
#else
#define USE_ASYNC_LDS 0
#endif

union FragU { v16bf v; __bf16 e[16]; };

// Load one 16x32 bf16 fragment from LDS (row-major tile, row stride LDS_LD).
// Layout per CDNA5 ISA 7.12.2 (16-bit A 16x32): lane = m + 16*khalf,
// VGPR pair j covers K = (j<4 ? 2j : 16+2(j-4)) + 8*khalf.  B tiles are staged
// transposed (NxK) so the same loader serves both operands.
__device__ inline v16bf load_frag16x32(const __bf16* base, int lane) {
  const int r  = lane & 15;
  const int kh = (lane >> 4) * 8;
  FragU f;
#pragma unroll
  for (int j = 0; j < 8; ++j) {
    const int kk = (j < 4 ? 2 * j : 16 + 2 * (j - 4)) + kh;
    f.e[2 * j]     = base[r * LDS_LD + kk];
    f.e[2 * j + 1] = base[r * LDS_LD + kk + 1];
  }
  return f.v;
}

// C = act(A[f32,MxK] @ Bt^T + bias); Bt is bf16 [N,K] row-major (pre-transposed
// weights).  act: 0 = none, 1 = exact GELU.  M%128==0, N%128==0, K%32==0.
__global__ __launch_bounds__(256)
void gemm_wmma_kernel(const float* __restrict__ A, const __bf16* __restrict__ Bt,
                      const float* __restrict__ bias, float* __restrict__ C,
                      int M, int Ncols, int K, int lda, int ldc, int act) {
  __shared__ __bf16 As[TILE_M][LDS_LD];
  __shared__ __bf16 Bs[TILE_N][LDS_LD];

  const int tid  = threadIdx.x;
  const int lane = tid & 31;
  const int wid  = tid >> 5;
  const int wm   = wid & 3;   // 4 waves along M  (4*32 = 128)
  const int wn   = wid >> 2;  // 2 waves along N  (2*64 = 128)
  const int blockM = blockIdx.y * TILE_M;
  const int blockN = blockIdx.x * TILE_N;

  v8f acc[2][4];
#pragma unroll
  for (int i = 0; i < 2; ++i)
#pragma unroll
    for (int j = 0; j < 4; ++j)
#pragma unroll
      for (int e = 0; e < 8; ++e) acc[i][j][e] = 0.0f;

  for (int k0 = 0; k0 < K; k0 += TILE_K) {
    // ---- stage B first: bf16 [N,K] tile straight into LDS -------------------
    {
      const int nn = tid >> 1;
      const int kb = (tid & 1) * 16;
      const __bf16* srcp = Bt + (size_t)(blockN + nn) * K + k0 + kb;
#if USE_ASYNC_LDS
      // gfx1250 GLOBAL_LOAD_ASYNC_TO_LDS_B128: per-lane 16B global->LDS DMA,
      // no VGPR round-trip, tracked by ASYNCcnt.
      __builtin_amdgcn_global_load_async_to_lds_b128(
          (gv4i_ptr)srcp, (lv4i_ptr)&Bs[nn][kb + 0], 0, 0);
      __builtin_amdgcn_global_load_async_to_lds_b128(
          (gv4i_ptr)(srcp + 8), (lv4i_ptr)&Bs[nn][kb + 8], 0, 0);
#else
      const uint4* src = reinterpret_cast<const uint4*>(srcp);
      *reinterpret_cast<uint4*>(&Bs[nn][kb + 0]) = src[0];
      *reinterpret_cast<uint4*>(&Bs[nn][kb + 8]) = src[1];
#endif
    }
    // ---- stage A (f32 -> bf16), 16 elems / thread, b128 global loads --------
    {
      const int row = tid >> 1;
      const int cb  = (tid & 1) * 16;
      const float* src = A + (size_t)(blockM + row) * lda + k0 + cb;
#pragma unroll
      for (int c = 0; c < 16; c += 4) {
        float4 vv = *reinterpret_cast<const float4*>(src + c);
        As[row][cb + c + 0] = (__bf16)vv.x;
        As[row][cb + c + 1] = (__bf16)vv.y;
        As[row][cb + c + 2] = (__bf16)vv.z;
        As[row][cb + c + 3] = (__bf16)vv.w;
      }
      if (k0 + TILE_K < K)  // gfx1250 global_prefetch of next A tile
        __builtin_prefetch(src + TILE_K, 0, 0);
    }
#if USE_ASYNC_LDS
    __builtin_amdgcn_s_wait_asynccnt(0);  // B tile resident in LDS
#endif
    __syncthreads();

    v16bf afr[2], bfr[4];
#pragma unroll
    for (int mt = 0; mt < 2; ++mt)
      afr[mt] = load_frag16x32(&As[wm * 32 + mt * 16][0], lane);
#pragma unroll
    for (int nt = 0; nt < 4; ++nt)
      bfr[nt] = load_frag16x32(&Bs[wn * 64 + nt * 16][0], lane);

#pragma unroll
    for (int mt = 0; mt < 2; ++mt)
#pragma unroll
      for (int nt = 0; nt < 4; ++nt)
        acc[mt][nt] = __builtin_amdgcn_wmma_f32_16x16x32_bf16(
            false, afr[mt], false, bfr[nt], (short)0, acc[mt][nt], false, false);
    __syncthreads();
  }

  // ---- epilogue: C/D layout -> m = e + 8*(lane>>4), n = lane&15 ----
  const int nloc = lane & 15;
  const int mhi  = (lane >> 4) * 8;
#pragma unroll
  for (int mt = 0; mt < 2; ++mt) {
#pragma unroll
    for (int nt = 0; nt < 4; ++nt) {
      const int gn  = blockN + wn * 64 + nt * 16 + nloc;
      const float bval = bias ? bias[gn] : 0.0f;
      const int gm0 = blockM + wm * 32 + mt * 16 + mhi;
#pragma unroll
      for (int e = 0; e < 8; ++e) {
        float x = acc[mt][nt][e] + bval;
        if (act == 1) x = 0.5f * x * (1.0f + erff(x * 0.70710678118654752f));
        C[(size_t)(gm0 + e) * ldc + gn] = x;
      }
    }
  }
}

// ---- tiled transpose + f32->bf16 convert: Bt[n][k] = (bf16)B[k][n] ----------
__global__ __launch_bounds__(256)
void transpose_cvt_kernel(const float* __restrict__ B, __bf16* __restrict__ Bt,
                          int K, int Ncols) {
  __shared__ float tile[32][33];
  const int kb = blockIdx.y * 32, nb = blockIdx.x * 32;
  const int tx = threadIdx.x & 31, ty = threadIdx.x >> 5;   // 32 x 8
#pragma unroll
  for (int r = 0; r < 32; r += 8)
    tile[ty + r][tx] = B[(size_t)(kb + ty + r) * Ncols + nb + tx];
  __syncthreads();
#pragma unroll
  for (int r = 0; r < 32; r += 8)
    Bt[(size_t)(nb + ty + r) * K + kb + tx] = (__bf16)tile[tx][ty + r];
}

// ---- LayerNorm, H == blockDim.x == 256, one block per row -------------------
__global__ __launch_bounds__(256)
void ln_kernel(const float* __restrict__ x, const float* __restrict__ g,
               const float* __restrict__ b, float* __restrict__ out, int H) {
  __shared__ float red[256];
  const int n = blockIdx.x, i = threadIdx.x;
  const float v = x[(size_t)n * H + i];
  red[i] = v; __syncthreads();
  for (int st = 128; st > 0; st >>= 1) { if (i < st) red[i] += red[i + st]; __syncthreads(); }
  const float mean = red[0] / (float)H; __syncthreads();
  const float d = v - mean;
  red[i] = d * d; __syncthreads();
  for (int st = 128; st > 0; st >>= 1) { if (i < st) red[i] += red[i + st]; __syncthreads(); }
  const float rstd = rsqrtf(red[0] / (float)H + 1e-5f);
  out[(size_t)n * H + i] = d * rstd * g[i] + b[i];
}

// ---- GLU (1024 -> 512) + LayerNorm, one block (256 thr) per row -------------
__global__ __launch_bounds__(256)
void glu_ln_kernel(const float* __restrict__ x, const float* __restrict__ g,
                   const float* __restrict__ b, float* __restrict__ out) {
  __shared__ float red[256];
  __shared__ float vals[512];
  const int n = blockIdx.x, t = threadIdx.x;
  const float* row = x + (size_t)n * 1024;
  float s = 0.0f;
#pragma unroll
  for (int u = 0; u < 2; ++u) {
    const int i = t + u * 256;
    const float a = row[i], bb = row[512 + i];
    const float gl = a / (1.0f + expf(-bb));
    vals[i] = gl; s += gl;
  }
  red[t] = s; __syncthreads();
  for (int st = 128; st > 0; st >>= 1) { if (t < st) red[t] += red[t + st]; __syncthreads(); }
  const float mean = red[0] * (1.0f / 512.0f); __syncthreads();
  float s2 = 0.0f;
#pragma unroll
  for (int u = 0; u < 2; ++u) { const int i = t + u * 256; const float d = vals[i] - mean; s2 += d * d; }
  red[t] = s2; __syncthreads();
  for (int st = 128; st > 0; st >>= 1) { if (t < st) red[t] += red[t + st]; __syncthreads(); }
  const float rstd = rsqrtf(red[0] * (1.0f / 512.0f) + 1e-5f);
#pragma unroll
  for (int u = 0; u < 2; ++u) {
    const int i = t + u * 256;
    out[(size_t)n * 512 + i] = (vals[i] - mean) * rstd * g[i] + b[i];
  }
}

// ---- init: act0 into pre[:,256:768], trace0 broadcast, ao0 ------------------
__global__ __launch_bounds__(256)
void ctm_init_kernel(float* __restrict__ pre, float* __restrict__ trace,
                     float* __restrict__ ao, const float* __restrict__ start_state,
                     const float* __restrict__ start_trace,
                     const int* __restrict__ idx_lo, const int* __restrict__ idx_ro) {
  const int idx = blockIdx.x * 256 + threadIdx.x;   // over N*512
  const int n = idx >> 9, d = idx & 511;
  pre[(size_t)n * 768 + 256 + d] = start_state[d];
#pragma unroll
  for (int m = 0; m < 8; ++m) trace[(size_t)idx * 8 + m] = start_trace[d * 8 + m];
  if (d < 64)
    ao[(size_t)n * 64 + d] = start_state[idx_lo[d]] * start_state[idx_ro[d]];
}

// ---- per-neuron NLM step: trace shift + GLU(8->16) + GLU(16->1) -------------
__global__ __launch_bounds__(256)
void ctm_step_kernel(const float* __restrict__ state, float* __restrict__ trace,
                     const float* __restrict__ w1, const float* __restrict__ b1,
                     const float* __restrict__ w2, const float* __restrict__ b2,
                     float* __restrict__ pre) {
  const int idx = blockIdx.x * 256 + threadIdx.x;   // over N*512
  const int n = idx >> 9, d = idx & 511;
  float* tp = trace + (size_t)idx * 8;
  float tr[8];
#pragma unroll
  for (int m = 0; m < 7; ++m) tr[m] = tp[m + 1];
  tr[7] = state[idx];
#pragma unroll
  for (int m = 0; m < 8; ++m) tp[m] = tr[m];

  float h1[16];
#pragma unroll
  for (int h = 0; h < 16; ++h) {
    float ua = b1[d * 32 + h], ub = b1[d * 32 + 16 + h];
#pragma unroll
    for (int m = 0; m < 8; ++m) {
      const float t = tr[m];
      ua += t * w1[((size_t)m * 512 + d) * 32 + h];
      ub += t * w1[((size_t)m * 512 + d) * 32 + 16 + h];
    }
    h1[h] = ua / (1.0f + expf(-ub));
  }
  float u0 = b2[d * 2 + 0], u1 = b2[d * 2 + 1];
#pragma unroll
  for (int h = 0; h < 16; ++h) {
    u0 += h1[h] * w2[((size_t)h * 512 + d) * 2 + 0];
    u1 += h1[h] * w2[((size_t)h * 512 + d) * 2 + 1];
  }
  pre[(size_t)n * 768 + 256 + d] = u0 / (1.0f + expf(-u1));
}

// ---- out-sync numerator recurrence ------------------------------------------
__global__ __launch_bounds__(256)
void sync_update_kernel(const float* __restrict__ pre, const int* __restrict__ idx_lo,
                        const int* __restrict__ idx_ro, const float* __restrict__ decay_out,
                        float* __restrict__ ao) {
  const int idx = blockIdx.x * 256 + threadIdx.x;   // over N*64
  const int n = idx >> 6, j = idx & 63;
  const float r = expf(-fminf(fmaxf(decay_out[j], 0.0f), 15.0f));
  const float al = pre[(size_t)n * 768 + 256 + idx_lo[j]];
  const float ar = pre[(size_t)n * 768 + 256 + idx_ro[j]];
  ao[idx] = r * ao[idx] + al * ar;
}

// ---- sync_final = ao * rsqrt(bb), bb via 25-step closed recurrence ----------
__global__ __launch_bounds__(256)
void finalize_sync_kernel(const float* __restrict__ ao, const float* __restrict__ decay_out,
                          float* __restrict__ syncf) {
  const int idx = blockIdx.x * 256 + threadIdx.x;   // over N*64
  const int j = idx & 63;
  const float r = expf(-fminf(fmaxf(decay_out[j], 0.0f), 15.0f));
  float bb = 1.0f;
#pragma unroll
  for (int t = 0; t < 25; ++t) bb = r * bb + 1.0f;
  syncf[idx] = ao[idx] * rsqrtf(bb);
}

// ---- certainty head: 64x64 projector, log_softmax, normalized entropy -------
__global__ __launch_bounds__(64)
void cert_kernel(const float* __restrict__ syncf, const float* __restrict__ op_w,
                 const float* __restrict__ op_b, float* __restrict__ cert) {
  __shared__ float sf[64];
  __shared__ float red[64];
  const int n = blockIdx.x, j = threadIdx.x;
  sf[j] = syncf[(size_t)n * 64 + j];
  __syncthreads();
  float acc = op_b[j];
  for (int k = 0; k < 64; ++k) acc += sf[k] * op_w[k * 64 + j];
  red[j] = acc; __syncthreads();
  for (int st = 32; st > 0; st >>= 1) { if (j < st) red[j] = fmaxf(red[j], red[j + st]); __syncthreads(); }
  const float mx = red[0]; __syncthreads();
  red[j] = expf(acc - mx); __syncthreads();
  for (int st = 32; st > 0; st >>= 1) { if (j < st) red[j] += red[j + st]; __syncthreads(); }
  const float lse = logf(red[0]); __syncthreads();
  const float logp = acc - mx - lse;
  red[j] = expf(logp) * logp; __syncthreads();
  for (int st = 32; st > 0; st >>= 1) { if (j < st) red[j] += red[j + st]; __syncthreads(); }
  if (j == 0) {
    const float ne = -red[0] * (1.0f / logf(64.0f));
    cert[(size_t)n * 2 + 0] = ne;
    cert[(size_t)n * 2 + 1] = 1.0f - ne;
  }
}

// ---------------------------------------------------------------------------
extern "C" void kernel_launch(void* const* d_in, const int* in_sizes, int n_in,
                              void* d_out, int out_size, void* d_ws, size_t ws_size,
                              hipStream_t stream) {
  (void)in_sizes; (void)n_in; (void)out_size; (void)ws_size;
  const int N = 1024, VOCAB = 50304;

  const float* hs       = (const float*)d_in[0];
  const float* in_w1    = (const float*)d_in[1];
  const float* in_b1    = (const float*)d_in[2];
  const float* in_w2    = (const float*)d_in[3];
  const float* in_b2    = (const float*)d_in[4];
  const float* in_ln_g  = (const float*)d_in[5];
  const float* in_ln_b  = (const float*)d_in[6];
  const float* kv_w     = (const float*)d_in[7];
  const float* kv_b     = (const float*)d_in[8];
  const float* kv_ln_g  = (const float*)d_in[9];
  const float* kv_ln_b  = (const float*)d_in[10];
  // d_in[11..16]: q_w,q_b,wq,bq,wk,bk -- dead (softmax over kv-length 1 == 1)
  const float* wv       = (const float*)d_in[17];
  const float* bv       = (const float*)d_in[18];
  const float* wo       = (const float*)d_in[19];
  const float* bo       = (const float*)d_in[20];
  const float* syn_w    = (const float*)d_in[21];
  const float* syn_b    = (const float*)d_in[22];
  const float* syn_ln_g = (const float*)d_in[23];
  const float* syn_ln_b = (const float*)d_in[24];
  const float* nlm_w1   = (const float*)d_in[25];
  const float* nlm_b1   = (const float*)d_in[26];
  const float* nlm_w2   = (const float*)d_in[27];
  const float* nlm_b2   = (const float*)d_in[28];
  const float* op_w     = (const float*)d_in[29];
  const float* op_b     = (const float*)d_in[30];
  const float* head_w1  = (const float*)d_in[31];
  const float* head_b1  = (const float*)d_in[32];
  const float* head_w2  = (const float*)d_in[33];
  const float* head_b2  = (const float*)d_in[34];
  const float* start_tr = (const float*)d_in[35];
  const float* start_st = (const float*)d_in[36];
  // d_in[37]: decay_action -- dead
  const float* decay_o  = (const float*)d_in[38];
  // d_in[39],[40]: idx_la, idx_ra -- dead
  const int* idx_lo     = (const int*)d_in[41];
  const int* idx_ro     = (const int*)d_in[42];

  // ---- workspace: f32 activations, then bf16 transposed weights ------------
  float* wsf = (float*)d_ws;
  size_t o = 0;
  float* t1     = wsf + o; o += (size_t)N * 512;
  float* t2     = wsf + o; o += (size_t)N * 256;
  float* feat   = wsf + o; o += (size_t)N * 256;
  float* kvr    = wsf + o; o += (size_t)N * 256;
  float* kv     = wsf + o; o += (size_t)N * 256;
  float* vbuf   = wsf + o; o += (size_t)N * 256;
  float* pre    = wsf + o; o += (size_t)N * 768;   // [attn | act]
  float* synout = wsf + o; o += (size_t)N * 1024;
  float* state  = wsf + o; o += (size_t)N * 512;
  float* trace  = wsf + o; o += (size_t)N * 512 * 8;
  float* ao     = wsf + o; o += (size_t)N * 64;
  float* syncf  = wsf + o; o += (size_t)N * 64;
  float* hmid   = wsf + o; o += (size_t)N * 768;

  __bf16* wsb = (__bf16*)(wsf + o);
  size_t ob = 0;
  __bf16* tw_in1 = wsb + ob; ob += (size_t)768 * 512;
  __bf16* tw_in2 = wsb + ob; ob += (size_t)512 * 256;
  __bf16* tw_kv  = wsb + ob; ob += (size_t)256 * 256;
  __bf16* tw_wv  = wsb + ob; ob += (size_t)256 * 256;
  __bf16* tw_wo  = wsb + ob; ob += (size_t)256 * 256;
  __bf16* tw_syn = wsb + ob; ob += (size_t)768 * 1024;
  __bf16* tw_h1  = wsb + ob; ob += (size_t)64 * 768;
  __bf16* tw_h2  = wsb + ob; ob += (size_t)768 * VOCAB;

  float* logits_out = (float*)d_out;
  float* cert_out   = (float*)d_out + (size_t)N * VOCAB;

  auto xpose = [&](const float* B, __bf16* Bt, int K, int Nc) {
    dim3 grid(Nc / 32, K / 32);
    transpose_cvt_kernel<<<grid, 256, 0, stream>>>(B, Bt, K, Nc);
  };
  auto gemm = [&](const float* A, const __bf16* Bt, const float* bias, float* C,
                  int M, int Nc, int K, int lda, int ldc, int act) {
    dim3 grid(Nc / TILE_N, M / TILE_M);
    gemm_wmma_kernel<<<grid, 256, 0, stream>>>(A, Bt, bias, C, M, Nc, K, lda, ldc, act);
  };

  // ---- weight transpose + bf16 convert (once per call) ---------------------
  xpose(in_w1, tw_in1, 768, 512);
  xpose(in_w2, tw_in2, 512, 256);
  xpose(kv_w,  tw_kv,  256, 256);
  xpose(wv,    tw_wv,  256, 256);
  xpose(wo,    tw_wo,  256, 256);
  xpose(syn_w, tw_syn, 768, 1024);
  xpose(head_w1, tw_h1, 64, 768);
  xpose(head_w2, tw_h2, 768, VOCAB);

  // in_proj: LN(gelu(hs@in_w1+b1)@in_w2+b2)
  gemm(hs, tw_in1, in_b1, t1, N, 512, 768, 768, 512, 1);
  gemm(t1, tw_in2, in_b2, t2, N, 256, 512, 512, 256, 0);
  ln_kernel<<<N, 256, 0, stream>>>(t2, in_ln_g, in_ln_b, feat, 256);
  // kv = LN(feat@kv_w+b)
  gemm(feat, tw_kv, kv_b, kvr, N, 256, 256, 256, 256, 0);
  ln_kernel<<<N, 256, 0, stream>>>(kvr, kv_ln_g, kv_ln_b, kv, 256);
  // attn = (kv@wv+bv)@wo+bo  (constant across iterations), into pre[:, :256]
  gemm(kv, tw_wv, bv, vbuf, N, 256, 256, 256, 256, 0);
  gemm(vbuf, tw_wo, bo, pre, N, 256, 256, 256, 768, 0);
  // CTM state init
  ctm_init_kernel<<<(N * 512) / 256, 256, 0, stream>>>(pre, trace, ao, start_st,
                                                       start_tr, idx_lo, idx_ro);
  // 25 CTM iterations
  for (int it = 0; it < 25; ++it) {
    gemm(pre, tw_syn, syn_b, synout, N, 1024, 768, 768, 1024, 0);
    glu_ln_kernel<<<N, 256, 0, stream>>>(synout, syn_ln_g, syn_ln_b, state);
    ctm_step_kernel<<<(N * 512) / 256, 256, 0, stream>>>(state, trace, nlm_w1, nlm_b1,
                                                         nlm_w2, nlm_b2, pre);
    sync_update_kernel<<<(N * 64) / 256, 256, 0, stream>>>(pre, idx_lo, idx_ro, decay_o, ao);
  }
  finalize_sync_kernel<<<(N * 64) / 256, 256, 0, stream>>>(ao, decay_o, syncf);
  cert_kernel<<<N, 64, 0, stream>>>(syncf, op_w, op_b, cert_out);
  // output head: gelu(syncf@head_w1+b1)@head_w2+b2
  gemm(syncf, tw_h1, head_b1, hmid, N, 768, 64, 64, 768, 1);
  gemm(hmid, tw_h2, head_b2, logits_out, N, VOCAB, 768, 768, VOCAB, 0);
}
